// StateSpaceBaseline_2534030705265
// MI455X (gfx1250) — matrix operand order repaired and checked by reading, not compiled, not run on hardware
//
#include <hip/hip_runtime.h>
#include <cmath>

// Problem constants (from reference)
#define B_   4096
#define T_   512
#define F_   32
#define S_   64
#define H_   128
#define OUT_ 32
#define PH_  6

// LDS tile row stride in halves: 136 halves = 272 B => bank advance of 4 per row,
// so a 32-lane ds_load_b128 A-fragment gather touches all 64 banks exactly twice
// (conflict-free).  (A stride of 128 halves = 256 B would alias every row onto the
// same banks: 16-way conflict on the scan critical path.)
#define LDS_STRIDE 136

typedef __attribute__((ext_vector_type(16))) _Float16 v16h;
typedef __attribute__((ext_vector_type(8)))  _Float16 v8h;
typedef __attribute__((ext_vector_type(8)))  float    v8f;

__device__ __forceinline__ v8f wmma_f16(v16h a, v16h b, v8f c) {
    // D = A(16x32 f16) * B(32x16 f16) + C(16x16 f32)
    return __builtin_amdgcn_wmma_f32_16x16x32_f16(
        /*neg_a=*/false, a, /*neg_b=*/false, b,
        /*c_mod=*/(short)0, c, /*reuse_a=*/false, /*reuse_b=*/false);
}

__device__ __forceinline__ float fast_tanh(float x) {
#if __has_builtin(__builtin_amdgcn_tanhf)
    return __builtin_amdgcn_tanhf(x);          // v_tanh_f32 (gfx1250 TRANS op)
#elif __has_builtin(__builtin_amdgcn_exp2f) && __has_builtin(__builtin_amdgcn_rcpf)
    float e = __builtin_amdgcn_exp2f(x * 2.885390081777926815f);
    return 1.0f - 2.0f * __builtin_amdgcn_rcpf(e + 1.0f);
#else
    return tanhf(x);
#endif
}

__device__ __forceinline__ v8f splat8(float s) {
    v8f r;
#pragma unroll
    for (int i = 0; i < 8; ++i) r[i] = s;
    return r;
}

__device__ __forceinline__ v16h pack16(float4 a, float4 b, float4 c, float4 d) {
    v16h f;
    f[0]=(_Float16)a.x; f[1]=(_Float16)a.y; f[2]=(_Float16)a.z; f[3]=(_Float16)a.w;
    f[4]=(_Float16)b.x; f[5]=(_Float16)b.y; f[6]=(_Float16)b.z; f[7]=(_Float16)b.w;
    f[8]=(_Float16)c.x; f[9]=(_Float16)c.y; f[10]=(_Float16)c.z; f[11]=(_Float16)c.w;
    f[12]=(_Float16)d.x; f[13]=(_Float16)d.y; f[14]=(_Float16)d.z; f[15]=(_Float16)d.w;
    return f;
}

// B-operand fragment for D = X @ W^T:  B[k, n] = W[n, k].
// f16 B 32x16 fragment, contiguous k order as in A: lane holds column n = lane&15; halves h=0..15 hold
// k = k0 + h where k0 = 32*slice + 16*(lane>=16).  16 consecutive floats of row n.
__device__ __forceinline__ v16h load_bfrag(const float* __restrict__ W,
                                           int rowLen, int n, int k0) {
    const float4* q = (const float4*)(W + (size_t)n * rowLen + k0);
    return pack16(q[0], q[1], q[2], q[3]);
}

// A-operand fragment (16x32 f16): lane holds row m = lane&15;
// halves 0..15 -> k = 16*hi + {0..15}.
__device__ __forceinline__ v16h load_x_afrag(const float* __restrict__ xt, int hi) {
    const float4* p0 = (const float4*)(xt + 16 * hi);
    return pack16(p0[0], p0[1], p0[2], p0[3]);
}

// Same A-fragment pattern, gathered from the f16 LDS tile.
__device__ __forceinline__ v16h lds_afrag(const _Float16* sm, int m, int kbase, int hi) {
    int k0 = kbase + 16 * hi;
    v8h lo = *(const v8h*)(sm + m * LDS_STRIDE + k0);        // ds_load_b128
    v8h hh = *(const v8h*)(sm + m * LDS_STRIDE + k0 + 8);    // ds_load_b128
    v16h f;
#pragma unroll
    for (int i = 0; i < 8; ++i) { f[i] = lo[i]; f[i + 8] = hh[i]; }
    return f;
}

__global__ __launch_bounds__(32) void ssm_scan_fused(
    const float* __restrict__ x,      const float* __restrict__ pl,
    const float* __restrict__ W_A,    const float* __restrict__ W_B,
    const float* __restrict__ W_C,    const float* __restrict__ b_C,
    const float* __restrict__ W_ctrl, const float* __restrict__ b_ctrl,
    const float* __restrict__ W1,     const float* __restrict__ b1,
    const float* __restrict__ W2,     const float* __restrict__ b2,
    float* __restrict__ out)
{
    __shared__ __attribute__((aligned(16))) _Float16 sm[16 * LDS_STRIDE];

    const int lane = threadIdx.x;
    const int l15  = lane & 15;
    const int hi   = lane >> 4;
    const int mb   = blockIdx.x * 16;     // batch-row base of this wave's tile

    // Zero the LDS state tile (state0 = 0).  16*136 halves = 272 uint4.
    {
        uint4 z = make_uint4(0u, 0u, 0u, 0u);
        uint4* p = (uint4*)sm;
        for (int i = lane; i < (16 * LDS_STRIDE * 2) / 16; i += 32) p[i] = z;
    }

    // Pre-pack W_A^T (4 N-tiles x 2 K-slices) and W_B^T (4 N-tiles) B-fragments.
    v16h wa[4][2], wb[4];
#pragma unroll
    for (int u = 0; u < 4; ++u) {
        int n = 16 * u + l15;
#pragma unroll
        for (int s = 0; s < 2; ++s)
            wa[u][s] = load_bfrag(W_A, S_, n, 32 * s + 16 * hi);
        wb[u] = load_bfrag(W_B, F_, n, 16 * hi);
    }

    // control[m, n] = b_ctrl[n] + pl[m,:] . W_ctrl[n,:]  in C/D accumulator layout:
    // tile u -> col n = 16u + l15;  vgpr v -> row m = v + 8*hi.
    float ctrl[4][8];
    {
        float plv[8][PH_];
#pragma unroll
        for (int v = 0; v < 8; ++v) {
            const float* pr = pl + (size_t)(mb + v + 8 * hi) * PH_;
#pragma unroll
            for (int p = 0; p < PH_; ++p) plv[v][p] = pr[p];
        }
#pragma unroll
        for (int u = 0; u < 4; ++u) {
            int n = 16 * u + l15;
            const float* wr = W_ctrl + (size_t)n * PH_;
            float w[PH_];
#pragma unroll
            for (int p = 0; p < PH_; ++p) w[p] = wr[p];
            float bc = b_ctrl[n];
#pragma unroll
            for (int v = 0; v < 8; ++v) {
                float s = bc;
#pragma unroll
                for (int p = 0; p < PH_; ++p) s += plv[v][p] * w[p];
                ctrl[u][v] = s;
            }
        }
    }

    const float* xrow = x + (size_t)(mb + l15) * T_ * F_;

    // ---- sequential scan over T ----
    // Software-pipeline the x fragment by one step; x-WMMAs go FIRST in each
    // step's chain so only 2 WMMAs sit between state-ready and tanh.
    v16h ax = load_x_afrag(xrow, hi);
    for (int t = 0; t < T_; ++t) {
        // Prefetch ~16 steps (2KB/row) ahead into L2/WGP$ (speculative => OOB-safe,
        // but clamp anyway).  hi=0 lanes cover bytes 0-63, hi=1 lanes 64-127.
        int tp = t + 16; if (tp >= T_) tp = T_ - 1;
        __builtin_prefetch(xrow + tp * F_ + 16 * hi, 0, 3);

        int tn = t + 1; if (tn >= T_) tn = T_ - 1;
        v16h ax_next = load_x_afrag(xrow + tn * F_, hi);

        v16h s0 = lds_afrag(sm, l15, 0,  hi);   // state K-slice 0..31
        v16h s1 = lds_afrag(sm, l15, 32, hi);   // state K-slice 32..63

        v8f acc[4];
#pragma unroll
        for (int u = 0; u < 4; ++u) {           // input-projection pass (A = ax)
#pragma unroll
            for (int v = 0; v < 8; ++v) acc[u][v] = ctrl[u][v];
            acc[u] = wmma_f16(ax, wb[u], acc[u]);
        }
#pragma unroll
        for (int u = 0; u < 4; ++u)             // recurrence pass, K-slice 0 (A = s0)
            acc[u] = wmma_f16(s0, wa[u][0], acc[u]);
#pragma unroll
        for (int u = 0; u < 4; ++u)             // recurrence pass, K-slice 1 (A = s1)
            acc[u] = wmma_f16(s1, wa[u][1], acc[u]);

#pragma unroll
        for (int u = 0; u < 4; ++u) {
#pragma unroll
            for (int v = 0; v < 8; ++v) {
                float z = fast_tanh(acc[u][v]);
                sm[(v + 8 * hi) * LDS_STRIDE + 16 * u + l15] = (_Float16)z;
            }
        }
        ax = ax_next;
        // per-wave LDS ops are in-order: next iteration's loads see these stores
    }

    // ---- head: obs = state@W_C^T + b_C ----
    v16h sa0 = lds_afrag(sm, l15, 0,  hi);
    v16h sa1 = lds_afrag(sm, l15, 32, hi);
#pragma unroll
    for (int u = 0; u < 8; ++u) {
        int n = 16 * u + l15;
        v8f o = splat8(b_C[n]);
        o = wmma_f16(sa0, load_bfrag(W_C, S_, n, 0  + 16 * hi), o);
        o = wmma_f16(sa1, load_bfrag(W_C, S_, n, 32 + 16 * hi), o);
#pragma unroll
        for (int v = 0; v < 8; ++v)
            sm[(v + 8 * hi) * LDS_STRIDE + n] = (_Float16)o[v];
    }

    // ---- h = relu(obs@W1^T + b1) ----
    v16h oa[4];
#pragma unroll
    for (int s = 0; s < 4; ++s) oa[s] = lds_afrag(sm, l15, 32 * s, hi);
#pragma unroll
    for (int u = 0; u < 8; ++u) {
        int n = 16 * u + l15;
        v8f h = splat8(b1[n]);
#pragma unroll
        for (int s = 0; s < 4; ++s)
            h = wmma_f16(oa[s], load_bfrag(W1, H_, n, 32 * s + 16 * hi), h);
#pragma unroll
        for (int v = 0; v < 8; ++v) {
            float z = h[v] > 0.0f ? h[v] : 0.0f;
            sm[(v + 8 * hi) * LDS_STRIDE + n] = (_Float16)z;
        }
    }

    // ---- out = h@W2^T + b2 ----
    v16h ha[4];
#pragma unroll
    for (int s = 0; s < 4; ++s) ha[s] = lds_afrag(sm, l15, 32 * s, hi);
#pragma unroll
    for (int u = 0; u < 2; ++u) {
        int n = 16 * u + l15;
        v8f o = splat8(b2[n]);
#pragma unroll
        for (int s = 0; s < 4; ++s)
            o = wmma_f16(ha[s], load_bfrag(W2, H_, n, 32 * s + 16 * hi), o);
#pragma unroll
        for (int v = 0; v < 8; ++v)
            out[(size_t)(mb + v + 8 * hi) * OUT_ + n] = o[v];
    }
}

extern "C" void kernel_launch(void* const* d_in, const int* in_sizes, int n_in,
                              void* d_out, int out_size, void* d_ws, size_t ws_size,
                              hipStream_t stream) {
    (void)in_sizes; (void)n_in; (void)out_size; (void)d_ws; (void)ws_size;
    const float* x      = (const float*)d_in[0];
    const float* pl     = (const float*)d_in[1];
    const float* W_A    = (const float*)d_in[2];
    const float* W_B    = (const float*)d_in[3];
    const float* W_C    = (const float*)d_in[4];
    const float* b_C    = (const float*)d_in[5];
    const float* W_ctrl = (const float*)d_in[6];
    const float* b_ctrl = (const float*)d_in[7];
    const float* W1     = (const float*)d_in[8];
    const float* b1     = (const float*)d_in[9];
    const float* W2     = (const float*)d_in[10];
    const float* b2     = (const float*)d_in[11];
    float* out = (float*)d_out;

    dim3 grid(B_ / 16);   // 256 single-wave workgroups (one 16-row batch tile each)
    dim3 block(32);       // wave32
    ssm_scan_fused<<<grid, block, 0, stream>>>(x, pl, W_A, W_B, W_C, b_C,
                                               W_ctrl, b_ctrl, W1, b1, W2, b2, out);
}
